// New_AE_78434692759784
// MI455X (gfx1250) — matrix operand order repaired and checked by reading, not compiled, not run on hardware
//
#include <hip/hip_runtime.h>
#include <hip/hip_bf16.h>
#include <stdint.h>

typedef __attribute__((ext_vector_type(16))) _Float16 v16h;
typedef __attribute__((ext_vector_type(8)))  _Float16 v8h;
typedef __attribute__((ext_vector_type(8)))  float    v8f;

constexpr int BSZ = 65536;
constexpr int M   = BSZ * 4;     // 262144 rows (b*4 + p)
constexpr int KD  = 128;
constexpr float INV_CNT = 1.0f / 4096.0f;   // 1024 batch * 4 positions per group
constexpr float EPS = 1e-5f;

// output section offsets (floats)
constexpr size_t O0 = 0;                 // jPxPyPzE  (B,4,4)
constexpr size_t O1 = (size_t)BSZ * 16;  // rec_jPxPyPzE
constexpr size_t O2 = (size_t)BSZ * 32;  // j_rot
constexpr size_t O3 = (size_t)BSZ * 48;  // rec_j (B,3,4)
constexpr size_t O4 = (size_t)BSZ * 60;  // z (B,16,4)

// workspace layout (bytes)
constexpr size_t OFF_X0  = 0;
constexpr size_t OFF_XA  = OFF_X0 + (size_t)M * 8;     // f16 [M,4]
constexpr size_t OFF_XB  = OFF_XA + (size_t)M * 256;   // f16 [M,128]
constexpr size_t OFF_XC  = OFF_XB + (size_t)M * 256;   // f16 [M,128]
constexpr size_t OFF_Y   = OFF_XC + (size_t)M * 32;    // f16 [M,16]
constexpr size_t OFF_SUM = OFF_Y  + (size_t)M * 512;   // f32 [M,128]
constexpr size_t OFF_WH  = OFF_SUM + 16384 * 4;        // SUM(8192 f32) + SQ(8192 f32)
// f16 weight copies: WEC @ +0 (32KB), WDC @ +32KB (32KB), WBI @ +64KB (4KB)

__device__ __forceinline__ float silu(float x) { return x / (1.0f + expf(-x)); }

__device__ __forceinline__ void async_copy16(unsigned lds_addr, const void* gaddr) {
  // CDNA5: per-lane async global->LDS 16B copy, tracked with ASYNCcnt.
  asm volatile("global_load_async_to_lds_b128 %0, %1, off"
               :: "v"(lds_addr), "v"((unsigned long long)(uintptr_t)gaddr)
               : "memory");
}
__device__ __forceinline__ void wait_async0() {
  asm volatile("s_wait_asynccnt 0" ::: "memory");
}

// ---------------------------------------------------------------- utilities
__global__ void zero_f32(float* __restrict__ p, int n) {
  int i = blockIdx.x * 256 + threadIdx.x;
  if (i < n) p[i] = 0.0f;
}

__global__ void cvt_f32_f16(const float* __restrict__ src, _Float16* __restrict__ dst, int n) {
  int i = blockIdx.x * 256 + threadIdx.x;
  if (i < n) dst[i] = (_Float16)src[i];
}

// ---------------------------------------------------------------- prep
__global__ __launch_bounds__(256) void prep_kernel(const float* __restrict__ j,
                                                   float* __restrict__ out,
                                                   _Float16* __restrict__ X0) {
  int idx = blockIdx.x * 256 + threadIdx.x;     // one (b,p)
  int b = idx >> 2, p = idx & 3;
  const float* jb = j + (size_t)b * 16;
  float pt = jb[p], eta = jb[4 + p], phi = jb[8 + p], mm = jb[12 + p];
  float ch = coshf(eta);
  float Px = pt * cosf(phi);
  float Py = pt * sinf(phi);
  float Pz = pt * sinhf(eta);
  float E  = sqrtf(pt * ch * pt * ch + mm * mm);
  float* o0 = out + O0 + (size_t)b * 16;
  o0[p] = Px; o0[4 + p] = Py; o0[8 + p] = Pz; o0[12 + p] = E;
  float* o2 = out + O2 + (size_t)b * 16;
  o2[p] = pt; o2[4 + p] = eta; o2[8 + p] = phi; o2[12 + p] = mm;
  _Float16* x = X0 + (size_t)idx * 4;
  x[0] = (_Float16)log1pf(pt);
  x[1] = (_Float16)eta;
  x[2] = (_Float16)phi;
  x[3] = (_Float16)log1pf(mm);
}

// ---------------------------------------------------------------- WMMA GEMM + GBN partial stats
// X: f16 [M,128] row-major. Wh: f16 [N,128] (pre-converted). Async-staged to LDS.
// Each block: 128 rows x N cols. 8 waves, wave w owns rows w*16..w*16+15.
template <int NT>
__global__ __launch_bounds__(256) void wmma_gemm_gbn(const _Float16* __restrict__ X,
                                                     const _Float16* __restrict__ Wh,
                                                     const float* __restrict__ bias,
                                                     float* __restrict__ Y,
                                                     float* __restrict__ gsum,
                                                     float* __restrict__ gsq) {
  constexpr int N = NT * 16;
  __shared__ __align__(32) _Float16 Xs[128 * KD];   // 32 KB
  __shared__ __align__(32) _Float16 Ws[N * KD];
  __shared__ float ssum[N];
  __shared__ float ssq[N];

  int tid = threadIdx.x;
  int rowbase = blockIdx.x * 128;

  // stage W tile into LDS via async DMA (16B per lane-op)
  {
    unsigned lbase = (unsigned)(uintptr_t)(void*)&Ws[0];
    const char* gw = (const char*)Wh;
    for (int i = tid; i < N * 16; i += 256)
      async_copy16(lbase + (unsigned)i * 16u, gw + (size_t)i * 16);
  }
  // stage X tile (128x128 halves = 32KB) via async DMA
  {
    unsigned lbase = (unsigned)(uintptr_t)(void*)&Xs[0];
    const char* gx = (const char*)(X + (size_t)rowbase * KD);
    #pragma unroll
    for (int i = 0; i < 8; ++i) {
      unsigned off = (unsigned)(tid * 16 + i * 4096);
      async_copy16(lbase + off, gx + off);
    }
  }
  if (tid < N) { ssum[tid] = 0.0f; ssq[tid] = 0.0f; }
  wait_async0();
  __syncthreads();

  int wave = tid >> 5;
  int lane = tid & 31;
  int hi   = lane >> 4;          // half-lane select
  int l15  = lane & 15;

  v8f acc[NT] = {};
  union AFrag { v16h v; v8h h[2]; };

  #pragma unroll
  for (int ks = 0; ks < 4; ++ks) {
    // A fragment: row = M index, 16 contiguous K halves (matches B)
    int row = wave * 16 + l15;
    int kb  = ks * 32 + hi * 16;
    AFrag A;
    A.h[0] = *(const v8h*)&Xs[row * KD + kb];
    A.h[1] = *(const v8h*)&Xs[row * KD + kb + 8];
    #pragma unroll
    for (int t = 0; t < NT; ++t) {
      int n   = t * 16 + l15;
      int kb2 = ks * 32 + hi * 16;                  // 16 contiguous K halves
      v16h Bf = *(const v16h*)&Ws[n * KD + kb2];
      acc[t] = __builtin_amdgcn_wmma_f32_16x16x32_f16(
          false, A.v, false, Bf, (short)0, acc[t], false, false);
    }
  }

  // epilogue: bias, store pre-norm Y, accumulate group stats
  int g = rowbase >> 12;
  int mofs = hi * 8;
  #pragma unroll
  for (int t = 0; t < NT; ++t) {
    int n = t * 16 + l15;
    float bv = bias[n];
    float ps = 0.0f, pq = 0.0f;
    #pragma unroll
    for (int r = 0; r < 8; ++r) {
      float y = acc[t][r] + bv;
      int m = rowbase + wave * 16 + r + mofs;
      Y[(size_t)m * N + n] = y;
      ps += y; pq += y * y;
    }
    atomicAdd(&ssum[n], ps);
    atomicAdd(&ssq[n], pq);
  }
  __syncthreads();
  if (tid < N) {
    atomicAdd(&gsum[g * N + tid], ssum[tid]);
    atomicAdd(&gsq[g * N + tid], ssq[tid]);
  }
}

// ---------------------------------------------------------------- small-K GEMMs (VALU)
// ie: K=3 (padded to 4), N=128. 128 threads = channels, 32 rows per block.
__global__ __launch_bounds__(128) void ie_gemm(const _Float16* __restrict__ X0,
                                               const float* __restrict__ w,
                                               const float* __restrict__ b,
                                               float* __restrict__ Y,
                                               float* __restrict__ gsum,
                                               float* __restrict__ gsq) {
  int n = threadIdx.x;
  int r0 = blockIdx.x * 32;
  int g = r0 >> 12;
  float w0 = w[n * 3], w1 = w[n * 3 + 1], w2 = w[n * 3 + 2], bv = b[n];
  float ps = 0.0f, pq = 0.0f;
  for (int r = 0; r < 32; ++r) {
    int m = r0 + r;
    const _Float16* x = X0 + (size_t)m * 4;
    float y = bv + w0 * (float)x[0] + w1 * (float)x[1] + w2 * (float)x[2];
    Y[(size_t)m * 128 + n] = y;
    ps += y; pq += y * y;
  }
  atomicAdd(&gsum[g * 128 + n], ps);
  atomicAdd(&gsq[g * 128 + n], pq);
}

// bo: K=16, N=128
__global__ __launch_bounds__(128) void bo_gemm(const _Float16* __restrict__ XC,
                                               const float* __restrict__ w,
                                               const float* __restrict__ b,
                                               float* __restrict__ Y,
                                               float* __restrict__ gsum,
                                               float* __restrict__ gsq) {
  __shared__ __align__(16) _Float16 xs[32 * 16];
  int tid = threadIdx.x;
  int r0 = blockIdx.x * 32;
  int g = r0 >> 12;
  ((uint64_t*)xs)[tid] = ((const uint64_t*)(XC + (size_t)r0 * 16))[tid];
  float wr[16];
  #pragma unroll
  for (int k = 0; k < 16; ++k) wr[k] = w[tid * 16 + k];
  float bv = b[tid];
  __syncthreads();
  float ps = 0.0f, pq = 0.0f;
  for (int r = 0; r < 32; ++r) {
    float y = bv;
    #pragma unroll
    for (int k = 0; k < 16; ++k) y += wr[k] * (float)xs[r * 16 + k];
    Y[(size_t)(r0 + r) * 128 + tid] = y;
    ps += y; pq += y * y;
  }
  atomicAdd(&gsum[g * 128 + tid], ps);
  atomicAdd(&gsq[g * 128 + tid], pq);
}

// or: K=128, N=3. 128 threads = rows.
__global__ __launch_bounds__(128) void or_gemm(const _Float16* __restrict__ XE,
                                               const float* __restrict__ w,
                                               const float* __restrict__ b,
                                               float* __restrict__ Y3,
                                               float* __restrict__ gsum,
                                               float* __restrict__ gsq) {
  __shared__ float ws[3 * 128];
  __shared__ float ss[6];
  int tid = threadIdx.x;
  for (int i = tid; i < 384; i += 128) ws[i] = w[i];
  if (tid < 6) ss[tid] = 0.0f;
  __syncthreads();
  int m = blockIdx.x * 128 + tid;
  int g = m >> 12;
  float a0 = b[0], a1 = b[1], a2 = b[2];
  const v8h* xv = (const v8h*)(XE + (size_t)m * 128);
  #pragma unroll
  for (int c = 0; c < 16; ++c) {
    v8h xc = xv[c];
    #pragma unroll
    for (int e = 0; e < 8; ++e) {
      float f = (float)xc[e];
      int k = c * 8 + e;
      a0 += ws[k] * f; a1 += ws[128 + k] * f; a2 += ws[256 + k] * f;
    }
  }
  Y3[(size_t)m * 3]     = a0;
  Y3[(size_t)m * 3 + 1] = a1;
  Y3[(size_t)m * 3 + 2] = a2;
  atomicAdd(&ss[0], a0); atomicAdd(&ss[1], a1); atomicAdd(&ss[2], a2);
  atomicAdd(&ss[3], a0 * a0); atomicAdd(&ss[4], a1 * a1); atomicAdd(&ss[5], a2 * a2);
  __syncthreads();
  if (tid < 3) {
    atomicAdd(&gsum[g * 3 + tid], ss[tid]);
    atomicAdd(&gsq[g * 3 + tid], ss[3 + tid]);
  }
}

// ---------------------------------------------------------------- GBN apply + SiLU (+residual, +z out)
__global__ __launch_bounds__(256) void apply_kernel(const float* __restrict__ Y,
                                                    const _Float16* __restrict__ Xres,
                                                    _Float16* __restrict__ Xout,
                                                    const float* __restrict__ gsum,
                                                    const float* __restrict__ gsq,
                                                    const float* __restrict__ gain,
                                                    const float* __restrict__ bias,
                                                    float* __restrict__ zout,
                                                    int nshift, int nmask) {
  size_t idx = (size_t)blockIdx.x * 256 + threadIdx.x;
  int n = (int)(idx & nmask);
  size_t m = idx >> nshift;
  int g = (int)(m >> 12);
  int N = nmask + 1;
  float mean = gsum[g * N + n] * INV_CNT;
  float var  = gsq[g * N + n] * INV_CNT - mean * mean;
  float inv  = rsqrtf(var + EPS);
  float xn = (Y[idx] - mean) * inv * gain[n] + bias[n];
  float s = silu(xn);
  if (Xres != nullptr) s += (float)Xres[idx];
  Xout[idx] = (_Float16)s;
  if (zout != nullptr) {
    size_t b = m >> 2; int p = (int)(m & 3);
    zout[b * 64 + (size_t)n * 4 + p] = s;
  }
}

// ---------------------------------------------------------------- final: or-apply + kinematics
__global__ __launch_bounds__(256) void apply_or_kernel(const float* __restrict__ Y3,
                                                       const float* __restrict__ gsum,
                                                       const float* __restrict__ gsq,
                                                       const float* __restrict__ gain,
                                                       const float* __restrict__ bias,
                                                       float* __restrict__ out) {
  int m = blockIdx.x * 256 + threadIdx.x;
  int g = m >> 12;
  float x[3];
  #pragma unroll
  for (int c = 0; c < 3; ++c) {
    float mean = gsum[g * 3 + c] * INV_CNT;
    float var  = gsq[g * 3 + c] * INV_CNT - mean * mean;
    float inv  = rsqrtf(var + EPS);
    float xn = (Y3[(size_t)m * 3 + c] - mean) * inv * gain[c] + bias[c];
    x[c] = silu(xn);
  }
  float rPt = coshf(x[0]) + 39.0f;
  float rEta = x[1], rPhi = x[2];
  size_t b = (size_t)(m >> 2); int p = m & 3;
  float* rj = out + O3 + b * 12;
  rj[p] = rPt; rj[4 + p] = rEta; rj[8 + p] = rPhi;
  float rPx = rPt * cosf(rPhi);
  float rPy = rPt * sinf(rPhi);
  float rPz = rPt * sinhf(rEta);
  float rE  = sqrtf(rPt * rPt + rPz * rPz);
  float* r4 = out + O1 + b * 16;
  r4[p] = rPx; r4[4 + p] = rPy; r4[8 + p] = rPz; r4[12 + p] = rE;
}

// ---------------------------------------------------------------- launcher
extern "C" void kernel_launch(void* const* d_in, const int* in_sizes, int n_in,
                              void* d_out, int out_size, void* d_ws, size_t ws_size,
                              hipStream_t stream) {
  (void)in_sizes; (void)n_in; (void)out_size; (void)ws_size;
  const float* j     = (const float*)d_in[0];
  const float* w_ie  = (const float*)d_in[1];
  const float* b_ie  = (const float*)d_in[2];
  const float* g_ie  = (const float*)d_in[3];
  const float* be_ie = (const float*)d_in[4];
  const float* w_ec  = (const float*)d_in[5];
  const float* b_ec  = (const float*)d_in[6];
  const float* g_ec  = (const float*)d_in[7];
  const float* be_ec = (const float*)d_in[8];
  const float* w_bi  = (const float*)d_in[9];
  const float* b_bi  = (const float*)d_in[10];
  const float* g_bi  = (const float*)d_in[11];
  const float* be_bi = (const float*)d_in[12];
  const float* w_bo  = (const float*)d_in[13];
  const float* b_bo  = (const float*)d_in[14];
  const float* g_bo  = (const float*)d_in[15];
  const float* be_bo = (const float*)d_in[16];
  const float* w_dc  = (const float*)d_in[17];
  const float* b_dc  = (const float*)d_in[18];
  const float* w_or  = (const float*)d_in[19];
  const float* b_or  = (const float*)d_in[20];
  const float* g_or  = (const float*)d_in[21];
  const float* be_or = (const float*)d_in[22];

  char* ws = (char*)d_ws;
  _Float16* X0 = (_Float16*)(ws + OFF_X0);
  _Float16* XA = (_Float16*)(ws + OFF_XA);   // also reused as XD
  _Float16* XB = (_Float16*)(ws + OFF_XB);   // also reused as XE
  _Float16* XC = (_Float16*)(ws + OFF_XC);
  float* Y   = (float*)(ws + OFF_Y);
  float* SUM = (float*)(ws + OFF_SUM);
  float* SQ  = SUM + 8192;
  _Float16* WEC = (_Float16*)(ws + OFF_WH);
  _Float16* WDC = WEC + 16384;
  _Float16* WBI = WDC + 16384;
  float* out = (float*)d_out;

  prep_kernel<<<M / 256, 256, 0, stream>>>(j, out, X0);
  cvt_f32_f16<<<64, 256, 0, stream>>>(w_ec, WEC, 16384);
  cvt_f32_f16<<<64, 256, 0, stream>>>(w_dc, WDC, 16384);
  cvt_f32_f16<<<8, 256, 0, stream>>>(w_bi, WBI, 2048);

  // L1: ie 3->128
  zero_f32<<<64, 256, 0, stream>>>(SUM, 16384);
  ie_gemm<<<M / 32, 128, 0, stream>>>(X0, w_ie, b_ie, Y, SUM, SQ);
  apply_kernel<<<(M / 256) * 128, 256, 0, stream>>>(Y, nullptr, XA, SUM, SQ, g_ie, be_ie, nullptr, 7, 127);

  // L2: ec 128->128 (WMMA) + residual
  zero_f32<<<64, 256, 0, stream>>>(SUM, 16384);
  wmma_gemm_gbn<8><<<M / 128, 256, 0, stream>>>(XA, WEC, b_ec, Y, SUM, SQ);
  apply_kernel<<<(M / 256) * 128, 256, 0, stream>>>(Y, XA, XB, SUM, SQ, g_ec, be_ec, nullptr, 7, 127);

  // L3: bi 128->16 (WMMA) + z output
  zero_f32<<<64, 256, 0, stream>>>(SUM, 16384);
  wmma_gemm_gbn<1><<<M / 128, 256, 0, stream>>>(XB, WBI, b_bi, Y, SUM, SQ);
  apply_kernel<<<(M / 256) * 16, 256, 0, stream>>>(Y, nullptr, XC, SUM, SQ, g_bi, be_bi, out + O4, 4, 15);

  // L4: bo 16->128
  zero_f32<<<64, 256, 0, stream>>>(SUM, 16384);
  bo_gemm<<<M / 32, 128, 0, stream>>>(XC, w_bo, b_bo, Y, SUM, SQ);
  apply_kernel<<<(M / 256) * 128, 256, 0, stream>>>(Y, nullptr, XA, SUM, SQ, g_bo, be_bo, nullptr, 7, 127);

  // L5: dc 128->128 (WMMA) + residual, reuses g_ec/be_ec
  zero_f32<<<64, 256, 0, stream>>>(SUM, 16384);
  wmma_gemm_gbn<8><<<M / 128, 256, 0, stream>>>(XA, WDC, b_dc, Y, SUM, SQ);
  apply_kernel<<<(M / 256) * 128, 256, 0, stream>>>(Y, XA, XB, SUM, SQ, g_ec, be_ec, nullptr, 7, 127);

  // L6: or 128->3 + final kinematics
  zero_f32<<<64, 256, 0, stream>>>(SUM, 16384);
  or_gemm<<<M / 128, 128, 0, stream>>>(XB, w_or, b_or, Y, SUM, SQ);
  apply_or_kernel<<<M / 256, 256, 0, stream>>>(Y, SUM, SQ, g_or, be_or, out);
}
